// DynamicGraphSpatialConv_30580167147628
// MI455X (gfx1250) — compile-verified, matrix operand order checked by reading, not run on hardware
//
#include <hip/hip_runtime.h>
#include <hip/hip_bf16.h>

// ---------------------------------------------------------------------------
// DynamicGraphSpatialConv collapsed to a single GEMM:
//   y[b,o,t] = sum_p G[o,p] * x[b,p,t] + bias_out[o],  p = cin*22 + m
// G (64 x 1408) is built once per launch by tiny prep kernels, split into
// bf16 hi/lo and pre-swizzled into WMMA A-fragment order. Main kernel does
// the bf16x3-precision GEMM with v_wmma_f32_16x16x32_bf16, f32 accumulate.
// Workspace requirement: ~1.5 MB.
// ---------------------------------------------------------------------------

typedef __attribute__((ext_vector_type(16))) __bf16 v16bf;
typedef __attribute__((ext_vector_type(8)))  float  v8f;

#define NB    64
#define NCIN  64
#define NCOUT 64
#define NNODE 22
#define NT    1024
#define KDIM  (NCIN * NNODE)   // 1408
#define NKT   (KDIM / 32)      // 44 k-tiles of 32

// workspace byte offsets
#define WS_T1   ((size_t)0)         // 484 f32
#define WS_T2   ((size_t)4096)      // 484 f32
#define WS_BIAS ((size_t)8192)      // 64 f32
#define WS_V    ((size_t)16384)     // 3*64*64*22 f32 = 1081344 B
#define WS_GH   ((size_t)1114112)   // 90112 bf16 (A-fragment swizzled)
#define WS_GL   ((size_t)1310720)   // 90112 bf16

__device__ __forceinline__ float sigm(float v) { return 1.0f / (1.0f + __expf(-v)); }

// ---------------------------------------------------------------------------
// Prep 1: dynamic adjacency -> scaled Laplacian S, T1=S, T2=2S^2-I, bias_out.
// One block, 64 threads (trivial work).
// ---------------------------------------------------------------------------
__global__ __launch_bounds__(64) void prep_small(
    const float* __restrict__ adj_param,
    const float* __restrict__ cheb_bias,
    const float* __restrict__ agg_w,     // [64][64][22]
    const float* __restrict__ agg_bias,
    float* __restrict__ T1, float* __restrict__ T2, float* __restrict__ biasws)
{
  __shared__ float adj_s[NNODE][NNODE];
  __shared__ float s_s[NNODE][NNODE];
  __shared__ float dis[NNODE];
  const int tid = threadIdx.x;

  for (int i = tid; i < NNODE * NNODE; i += 64) {
    const int a = i / NNODE, c = i % NNODE;
    float v = 0.5f * (sigm(adj_param[a * NNODE + c]) + sigm(adj_param[c * NNODE + a]));
    if (a == c) v = 0.0f;
    adj_s[a][c] = v;
  }
  __syncthreads();
  if (tid < NNODE) {
    float d = 0.0f;
    for (int j = 0; j < NNODE; ++j) d += adj_s[tid][j];
    dis[tid] = rsqrtf(d);
  }
  __syncthreads();
  for (int i = tid; i < NNODE * NNODE; i += 64) {
    const int a = i / NNODE, c = i % NNODE;
    const float sv = -dis[a] * adj_s[a][c] * dis[c];
    s_s[a][c] = sv;
    T1[i] = sv;
  }
  __syncthreads();
  for (int i = tid; i < NNODE * NNODE; i += 64) {
    const int a = i / NNODE, c = i % NNODE;
    float acc = 0.0f;
    for (int n = 0; n < NNODE; ++n) acc += s_s[a][n] * s_s[n][c];
    T2[i] = 2.0f * acc - (a == c ? 1.0f : 0.0f);
  }
  if (tid < NCOUT) {
    float bo = agg_bias[tid];
    for (int c = 0; c < NCOUT; ++c) {
      const float cb = cheb_bias[c];
      for (int n = 0; n < NNODE; ++n)
        bo += agg_w[(tid * NCOUT + c) * NNODE + n] * cb;
    }
    biasws[tid] = bo;
  }
}

// ---------------------------------------------------------------------------
// Prep 2: V[k][o][cin][v] = sum_cout agg_w[o,cout,v] * W_k[cin,cout]
// 270336 threads, 64 MACs each.
// ---------------------------------------------------------------------------
__global__ __launch_bounds__(256) void prep_V(
    const float* __restrict__ agg_w,   // [64][64][22]
    const float* __restrict__ cw,      // [3][64][64]
    float* __restrict__ V)
{
  const int idx = blockIdx.x * 256 + threadIdx.x;
  if (idx >= 3 * NCOUT * NCIN * NNODE) return;
  const int n   = idx % NNODE;
  int q         = idx / NNODE;
  const int cin = q % NCIN;  q /= NCIN;
  const int o   = q % NCOUT;
  const int k   = q / NCOUT;
  float acc = 0.0f;
  for (int cout = 0; cout < NCOUT; ++cout)
    acc += agg_w[(o * NCOUT + cout) * NNODE + n] * cw[(k * NCIN + cin) * NCOUT + cout];
  V[idx] = acc;
}

// ---------------------------------------------------------------------------
// Prep 3: G[o][p] = V0[o,cin,m] + sum_v V1[o,cin,v]*T1[v,m] + V2[o,cin,v]*T2[v,m]
// then bf16 hi/lo split, stored pre-swizzled in WMMA A-fragment order:
//   frag = kt*4 + mt ; lane = h*16 + r ; 16 bf16 per lane contiguous.
//   K(ko): h = (ko>>3)&1, j = ((ko>>4)<<3) + (ko&7)   (ISA 16-bit A 16x32 layout)
// ---------------------------------------------------------------------------
__global__ __launch_bounds__(256) void prep_G(
    const float* __restrict__ V,
    const float* __restrict__ T1, const float* __restrict__ T2,
    __bf16* __restrict__ Gh, __bf16* __restrict__ Gl)
{
  const int idx = blockIdx.x * 256 + threadIdx.x;
  if (idx >= NCOUT * KDIM) return;
  const int o   = idx / KDIM;
  const int p   = idx % KDIM;
  const int cin = p / NNODE;
  const int m   = p % NNODE;

  float g = V[((0 * NCOUT + o) * NCIN + cin) * NNODE + m];
  for (int v = 0; v < NNODE; ++v)
    g += V[((1 * NCOUT + o) * NCIN + cin) * NNODE + v] * T1[v * NNODE + m];
  for (int v = 0; v < NNODE; ++v)
    g += V[((2 * NCOUT + o) * NCIN + cin) * NNODE + v] * T2[v * NNODE + m];

  const __bf16 hi = (__bf16)g;
  const __bf16 lo = (__bf16)(g - (float)hi);

  // swizzle (o,p) -> fragment position
  const int kt = p >> 5, ko = p & 31;
  const int mt = o >> 4, r = o & 15;
  const int h = (ko >> 3) & 1;
  const int j = ((ko >> 4) << 3) + (ko & 7);
  const int lane = (h << 4) | r;
  const size_t pos = ((size_t)(kt * 4 + mt) * 32 + lane) * 16 + j;
  Gh[pos] = hi;
  Gl[pos] = lo;
}

// ---------------------------------------------------------------------------
// Main GEMM: y[b,o,t] = G x + bias. Each wave: 2 M-tiles x 4 N-tiles,
// full K loop (44 steps of 32). bf16x3: acc += Gl*xh + Gh*xl + Gh*xh.
// 2048 waves = 256 blocks x 8 waves.
// ---------------------------------------------------------------------------
__global__ __launch_bounds__(256) void dgsc_gemm(
    const float* __restrict__ x,
    const __bf16* __restrict__ Gh,
    const __bf16* __restrict__ Gl,
    const float* __restrict__ biasv,
    float* __restrict__ out)
{
  const int lane  = threadIdx.x & 31;
  const int wid   = threadIdx.x >> 5;
  const int gw    = blockIdx.x * 8 + wid;   // 0..2047
  const int b     = gw >> 5;                // 0..63
  const int rem   = gw & 31;
  const int mh    = rem >> 4;               // M-tile pair {2mh, 2mh+1}
  const int tq    = rem & 15;               // 64 t-columns per wave
  const int tbase = tq * 64;
  const int h     = lane >> 4;
  const int r     = lane & 15;

  const v8f vzero = {0.f, 0.f, 0.f, 0.f, 0.f, 0.f, 0.f, 0.f};
  v8f acc[2][4];
#pragma unroll
  for (int i = 0; i < 2; ++i)
#pragma unroll
    for (int nt = 0; nt < 4; ++nt) acc[i][nt] = vzero;

  const float* xb = x + (size_t)b * (NCIN * NNODE * NT);

  for (int kt = 0; kt < NKT; ++kt) {
    // A fragments: one contiguous 32B load per lane per (matrix, m-tile)
    v16bf ah[2], al[2];
#pragma unroll
    for (int i = 0; i < 2; ++i) {
      const int mt = 2 * mh + i;
      const size_t fo = ((size_t)(kt * 4 + mt) * 32 + lane) * 16;
      ah[i] = *(const v16bf*)(Gh + fo);
      al[i] = *(const v16bf*)(Gl + fo);
    }
    if (kt + 1 < NKT) {  // pull next G chunk toward L0 (global_prefetch_b8)
      const size_t fn = ((size_t)((kt + 1) * 4 + 2 * mh) * 32 + lane) * 16;
      __builtin_prefetch((const void*)(Gh + fn), 0, 0);
      __builtin_prefetch((const void*)(Gl + fn), 0, 0);
    }

#pragma unroll
    for (int nt = 0; nt < 4; ++nt) {
      // B fragment: lane (h,r) holds column t = tbase+nt*16+r,
      // K = kt*32 + {h*8+0..7, 16+h*8+0..7}; 16 coalesced b32 loads.
      const float* px = xb + (size_t)(kt * 32 + h * 8) * NT + (tbase + nt * 16 + r);
      float xv[16];
#pragma unroll
      for (int j = 0; j < 16; ++j) {
        const int krel = ((j >> 3) << 4) + (j & 7);
        xv[j] = px[(size_t)krel * NT];
      }
      v16bf bh, bl;
#pragma unroll
      for (int j = 0; j < 16; ++j) {
        const __bf16 hi = (__bf16)xv[j];
        bh[j] = hi;
        bl[j] = (__bf16)(xv[j] - (float)hi);
      }
#pragma unroll
      for (int i = 0; i < 2; ++i) {
        acc[i][nt] = __builtin_amdgcn_wmma_f32_16x16x32_bf16(
            false, al[i], false, bh, (short)0, acc[i][nt], false, false);
        acc[i][nt] = __builtin_amdgcn_wmma_f32_16x16x32_bf16(
            false, ah[i], false, bl, (short)0, acc[i][nt], false, false);
        acc[i][nt] = __builtin_amdgcn_wmma_f32_16x16x32_bf16(
            false, ah[i], false, bh, (short)0, acc[i][nt], false, false);
      }
    }
  }

  // Epilogue: D layout — VGPR v holds row M = mt*16 + h*8 + v, col N = r.
#pragma unroll
  for (int i = 0; i < 2; ++i) {
    const int mt = 2 * mh + i;
#pragma unroll
    for (int v = 0; v < 8; ++v) {
      const int o = mt * 16 + h * 8 + v;
      const float bo = biasv[o];
      float* po = out + ((size_t)b * NCOUT + o) * NT + tbase + r;
#pragma unroll
      for (int nt = 0; nt < 4; ++nt)
        po[nt * 16] = acc[i][nt][v] + bo;
    }
  }
}

// ---------------------------------------------------------------------------
extern "C" void kernel_launch(void* const* d_in, const int* in_sizes, int n_in,
                              void* d_out, int out_size, void* d_ws, size_t ws_size,
                              hipStream_t stream) {
  (void)in_sizes; (void)n_in; (void)out_size; (void)ws_size;
  const float* x        = (const float*)d_in[0];  // [64,64,22,1024]
  const float* adjp     = (const float*)d_in[1];  // [22,22]
  const float* cw       = (const float*)d_in[2];  // [3,64,64]
  const float* cb       = (const float*)d_in[3];  // [64]
  const float* aw       = (const float*)d_in[4];  // [64,64,22,1]
  const float* ab       = (const float*)d_in[5];  // [64]
  float* out            = (float*)d_out;          // [64,64,1,1024]

  char* ws = (char*)d_ws;
  float*  T1   = (float*)(ws + WS_T1);
  float*  T2   = (float*)(ws + WS_T2);
  float*  bws  = (float*)(ws + WS_BIAS);
  float*  V    = (float*)(ws + WS_V);
  __bf16* Gh   = (__bf16*)(ws + WS_GH);
  __bf16* Gl   = (__bf16*)(ws + WS_GL);

  prep_small<<<1, 64, 0, stream>>>(adjp, cb, aw, ab, T1, T2, bws);
  prep_V<<<(3 * NCOUT * NCIN * NNODE + 255) / 256, 256, 0, stream>>>(aw, cw, V);
  prep_G<<<(NCOUT * KDIM + 255) / 256, 256, 0, stream>>>(V, T1, T2, Gh, Gl);
  dgsc_gemm<<<256, 256, 0, stream>>>(x, Gh, Gl, bws, out);
}
